// PARMAImproved_1219770712151
// MI455X (gfx1250) — compile-verified, matrix operand order checked by reading, not run on hardware
//
#include <hip/hip_runtime.h>
#include <hip/hip_bf16.h>
#include <math.h>

typedef __attribute__((ext_vector_type(16))) __bf16        v16bf;
typedef __attribute__((ext_vector_type(8)))  float         v8f;
typedef __attribute__((ext_vector_type(8)))  unsigned int  v8u;

#define DIM 128

// ---------------- degree / edge-weight precompute ----------------

__global__ void k_deg_init(float* deg, int n) {
    int i = blockIdx.x * blockDim.x + threadIdx.x;
    if (i < n) deg[i] = 1.0f;   // self-loop contributes 1
}

__global__ void k_deg_count(const int* dst, float* deg, int e) {
    int i = blockIdx.x * blockDim.x + threadIdx.x;
    if (i < e) atomicAdd(&deg[dst[i]], 1.0f);
}

__global__ void k_rsqrt_inplace(float* deg, int n) {
    int i = blockIdx.x * blockDim.x + threadIdx.x;
    if (i < n) {
        float d = deg[i];
        deg[i] = d > 0.0f ? rsqrtf(d) : 0.0f;
    }
}

__global__ void k_edge_w(const int* src, const int* dst, const float* dinv,
                         float* ew, int e) {
    int i = blockIdx.x * blockDim.x + threadIdx.x;
    if (i < e) ew[i] = dinv[src[i]] * dinv[dst[i]];
}

// ---------------- propagate: self-loop init + edge scatter ----------------

__global__ void k_self(const float* __restrict__ S, const float* __restrict__ dinv,
                       float* __restrict__ Snew, int n) {
    int t = blockIdx.x * blockDim.x + threadIdx.x;
    int node = t >> 5;
    int f = (t & 31) * 4;
    if (node < n) {
        float w = dinv[node];
        w = w * w;
        float4 v = *(const float4*)(S + (size_t)node * DIM + f);
        float4 o;
        o.x = w * v.x; o.y = w * v.y; o.z = w * v.z; o.w = w * v.w;
        *(float4*)(Snew + (size_t)node * DIM + f) = o;
    }
}

// one wave per edge; lane l covers features [4l, 4l+4)
__global__ void k_scatter(const int* __restrict__ src, const int* __restrict__ dst,
                          const float* __restrict__ ew, const float* __restrict__ S,
                          float* __restrict__ Snew, int e) {
    int gid = blockIdx.x * blockDim.x + threadIdx.x;
    int edge = gid >> 5;
    int lane = gid & 31;
    if (edge < e) {
        int s = src[edge], d = dst[edge];
        float w = ew[edge];
        float4 v = *(const float4*)(S + (size_t)s * DIM + lane * 4);
        float* o = Snew + (size_t)d * DIM + lane * 4;
        atomicAdd(o + 0, w * v.x);
        atomicAdd(o + 1, w * v.y);
        atomicAdd(o + 2, w * v.z);
        atomicAdd(o + 3, w * v.w);
    }
}

// ---------------- B pre-pack into WMMA fragment layout ----------------
// ISA 16-bit operand layout: per-lane element pair j packs
//   K = kbase + (j<4 ? 2j : 16+2(j-4)) + (lane>>4)*8, and K+1.
// Packed layout: fragment (chunk, ntile, lane) = 16 bf16 = 32 bytes, stored at
//   out + ((chunk*8 + ntile)*32 + lane)*8  (uint units)
// so a wave's fragment load is 32 consecutive 32B blocks -> coalesced b128s.

__device__ __forceinline__ int kpat(int kbase, int j, int half) {
    return kbase + (j < 4 ? 2 * j : 16 + 2 * (j - 4)) + half * 8;
}

__global__ void k_pack_b(const float* __restrict__ B, int ldk, int ldn,
                         unsigned int* __restrict__ out) {
    int idx = blockIdx.x * blockDim.x + threadIdx.x;   // 0..1023
    if (idx >= 1024) return;
    int lane  = idx & 31;
    int nt    = (idx >> 5) & 7;
    int chunk = idx >> 8;                              // 0..3
    int half = lane >> 4, lm = lane & 15;
    int n = nt * 16 + lm;
    int kbase = chunk * 32;
    unsigned int frag[8];
#pragma unroll
    for (int j = 0; j < 8; ++j) {
        int kk = kpat(kbase, j, half);
        __bf16 lo = (__bf16)B[(size_t)kk * ldk + (size_t)n * ldn];
        __bf16 hi = (__bf16)B[(size_t)(kk + 1) * ldk + (size_t)n * ldn];
        unsigned int ulo = (unsigned int)__builtin_bit_cast(unsigned short, lo);
        unsigned int uhi = (unsigned int)__builtin_bit_cast(unsigned short, hi);
        frag[j] = ulo | (uhi << 16);
    }
    unsigned int* o = out + ((size_t)((chunk << 3) + nt) * 32 + lane) * 8;
    *(uint4*)(o)     = make_uint4(frag[0], frag[1], frag[2], frag[3]);
    *(uint4*)(o + 4) = make_uint4(frag[4], frag[5], frag[6], frag[7]);
}

// ---------------- WMMA GEMM: C = A1@B1 (+ A2@B2) (+bias) (+relu) ----------------

// A fragment for (kbase, half) = two contiguous 8-float runs:
//   run0 = row[kbase + 8*half .. +8)   -> a[0..7]
//   run1 = row[kbase + 16 + 8*half ..] -> a[8..15]
__device__ __forceinline__ v16bf load_a_chunk(const float* __restrict__ row,
                                              int kbase, int half) {
    v8f r0 = *(const v8f*)(row + kbase + 8 * half);
    v8f r1 = *(const v8f*)(row + kbase + 16 + 8 * half);
    v16bf a;
#pragma unroll
    for (int i = 0; i < 8; ++i) {
        a[i]     = (__bf16)r0[i];
        a[8 + i] = (__bf16)r1[i];
    }
    return a;
}

__device__ __forceinline__ v16bf load_b_frag(const unsigned int* __restrict__ Bpk,
                                             int chunk, int nt, int lane) {
    const v8u* p = (const v8u*)(Bpk + ((size_t)((chunk << 3) + nt) * 32 + lane) * 8);
    return __builtin_bit_cast(v16bf, *p);
}

// block = 256 threads = 8 waves; each wave produces 16 rows x 128 cols.
__global__ void __launch_bounds__(256)
k_gemm(const float* __restrict__ A1, const unsigned int* __restrict__ B1pk,
       const float* __restrict__ A2, const unsigned int* __restrict__ B2pk,
       const float* __restrict__ biasv, int do_relu,
       float* __restrict__ C, int nrows) {
    int wave = threadIdx.x >> 5;
    int lane = threadIdx.x & 31;
    int half = lane >> 4;
    int lm   = lane & 15;
    int row0 = blockIdx.x * 128 + wave * 16;

    int myM = row0 + lm;
    int mclamp = myM < nrows ? myM : (nrows - 1);

    v8f acc[8];
#pragma unroll
    for (int nt = 0; nt < 8; ++nt) acc[nt] = (v8f){0.f,0.f,0.f,0.f,0.f,0.f,0.f,0.f};

    {
        const float* arow = A1 + (size_t)mclamp * DIM;
#pragma unroll
        for (int c = 0; c < 4; ++c) {
            v16bf a = load_a_chunk(arow, c * 32, half);
#pragma unroll
            for (int nt = 0; nt < 8; ++nt) {
                v16bf b = load_b_frag(B1pk, c, nt, lane);
                acc[nt] = __builtin_amdgcn_wmma_f32_16x16x32_bf16(
                    false, a, false, b, (short)0, acc[nt], false, false);
            }
        }
    }
    if (A2 != nullptr) {
        const float* arow = A2 + (size_t)mclamp * DIM;
#pragma unroll
        for (int c = 0; c < 4; ++c) {
            v16bf a = load_a_chunk(arow, c * 32, half);
#pragma unroll
            for (int nt = 0; nt < 8; ++nt) {
                v16bf b = load_b_frag(B2pk, c, nt, lane);
                acc[nt] = __builtin_amdgcn_wmma_f32_16x16x32_bf16(
                    false, a, false, b, (short)0, acc[nt], false, false);
            }
        }
    }

    // C/D layout: acc[nt][r] -> row = row0 + r + half*8, col = nt*16 + lm
#pragma unroll
    for (int r = 0; r < 8; ++r) {
        int rr = row0 + half * 8 + r;
        if (rr < nrows) {
#pragma unroll
            for (int nt = 0; nt < 8; ++nt) {
                int col = nt * 16 + lm;
                float v = acc[nt][r];
                if (biasv) v += biasv[col];
                if (do_relu) v = fmaxf(v, 0.0f);
                C[(size_t)rr * DIM + col] = v;
            }
        }
    }
}

// ---------------- log_softmax over dim=128, wave per row ----------------

__global__ void k_logsoftmax(float* __restrict__ out, int n) {
    int gid = blockIdx.x * blockDim.x + threadIdx.x;
    int row = gid >> 5;
    int lane = gid & 31;
    if (row >= n) return;
    float4 v = *(const float4*)(out + (size_t)row * DIM + lane * 4);
    float m = fmaxf(fmaxf(v.x, v.y), fmaxf(v.z, v.w));
#pragma unroll
    for (int off = 16; off >= 1; off >>= 1) m = fmaxf(m, __shfl_xor(m, off, 32));
    float s = expf(v.x - m) + expf(v.y - m) + expf(v.z - m) + expf(v.w - m);
#pragma unroll
    for (int off = 16; off >= 1; off >>= 1) s += __shfl_xor(s, off, 32);
    float ls = m + logf(s);
    float4 o;
    o.x = v.x - ls; o.y = v.y - ls; o.z = v.z - ls; o.w = v.w - ls;
    *(float4*)(out + (size_t)row * DIM + lane * 4) = o;
}

// ---------------- driver ----------------

static inline int cdiv(long long a, int b) { return (int)((a + b - 1) / b); }

extern "C" void kernel_launch(void* const* d_in, const int* in_sizes, int n_in,
                              void* d_out, int out_size, void* d_ws, size_t ws_size,
                              hipStream_t stream) {
    const float* x       = (const float*)d_in[0];   // [N,128]
    const int*   ei      = (const int*)  d_in[1];   // [2,E]
    const float* weight  = (const float*)d_in[2];   // [K,128,128]
    const float* root_w  = (const float*)d_in[3];   // [K,128,128]
    const float* init_W  = (const float*)d_in[4];   // [128,128]
    const float* init_b  = (const float*)d_in[5];   // [128]
    const float* bias    = (const float*)d_in[6];   // [T,K,128]

    const int n = in_sizes[0] / DIM;
    const int e = in_sizes[1] / 2;
    const int K = in_sizes[2] / (DIM * DIM);
    const int T = in_sizes[6] / (K * DIM);

    const int* src = ei;
    const int* dst = ei + e;

    float* C    = (float*)d_out;                    // current activation (ping)
    float* dinv = (float*)d_ws;                     // [n]
    float* ew   = dinv + n;                         // [e]
    float* S    = ew + e;                           // [n,128] scatter buffer (pong)
    unsigned int* pk = (unsigned int*)(S + (size_t)n * DIM);  // packed B: (1+2K)*8192 uints

    const int thr = 256;
    const size_t PKSZ = 8192;                       // uints per packed 128x128 matrix

    // gcn_norm
    k_deg_init     <<<cdiv(n, thr), thr, 0, stream>>>(dinv, n);
    k_deg_count    <<<cdiv(e, thr), thr, 0, stream>>>(dst, dinv, e);
    k_rsqrt_inplace<<<cdiv(n, thr), thr, 0, stream>>>(dinv, n);
    k_edge_w       <<<cdiv(e, thr), thr, 0, stream>>>(src, dst, dinv, ew, e);

    // pack all weight matrices to bf16 WMMA fragment layout
    unsigned int* pk_init = pk;
    k_pack_b<<<4, thr, 0, stream>>>(init_W, 1, DIM, pk_init);   // B = init_W^T
    for (int k = 0; k < K; ++k) {
        k_pack_b<<<4, thr, 0, stream>>>(weight + (size_t)k * DIM * DIM, DIM, 1,
                                        pk + (1 + k) * PKSZ);
        k_pack_b<<<4, thr, 0, stream>>>(root_w + (size_t)k * DIM * DIM, DIM, 1,
                                        pk + (1 + K + k) * PKSZ);
    }

    const int gemm_blocks = cdiv(n, 128);

    // out = x @ init_W^T + init_b
    k_gemm<<<gemm_blocks, thr, 0, stream>>>(x, pk_init, nullptr, nullptr,
                                            init_b, 0, C, n);

    for (int t = 0; t < T; ++t) {
        for (int k = 0; k < K; ++k) {
            // propagate: S = D^-1/2 (A + I) D^-1/2 * C
            k_self   <<<cdiv((long long)n * 32, thr), thr, 0, stream>>>(C, dinv, S, n);
            k_scatter<<<cdiv((long long)e * 32, thr), thr, 0, stream>>>(src, dst, ew, C, S, e);

            const float* bp = (k == K - 1) ? (bias + ((size_t)t * K + (K - 1)) * DIM) : nullptr;
            int relu = (k == K - 1) ? 1 : 0;
            // C = S @ weight[k] + x @ root_weight[k] (+bias, +relu at k==K-1)
            k_gemm<<<gemm_blocks, thr, 0, stream>>>(
                S, pk + (1 + k) * PKSZ,
                x, pk + (1 + K + k) * PKSZ,
                bp, relu, C, n);
        }
    }

    k_logsoftmax<<<cdiv((long long)n * 32, thr), thr, 0, stream>>>(C, n);
}
